// KDA_63359357551384
// MI455X (gfx1250) — matrix operand order repaired and checked by reading, not compiled, not run on hardware
//
#include <hip/hip_runtime.h>
#include <hip/hip_bf16.h>
#include <stdint.h>
#include <stddef.h>

// ---------------- problem constants ----------------
#define T_     1024
#define DIM_   1024
#define H_     8
#define QLR_   768
#define KVLR_  512
#define DNOPE_ 128
#define DROPE_ 64
#define DV_    128
#define DK_    192   // DNOPE + DROPE
#define KVA_N  640   // KVLR + DROPE (=576) zero-padded to a multiple of 128

typedef __bf16 bf16;
typedef __attribute__((ext_vector_type(16))) __bf16 v16bf;
typedef __attribute__((ext_vector_type(8)))  float  v8f;
typedef __attribute__((ext_vector_type(4)))  int    v4i;

// CDNA5 async global->LDS staging (ASYNCcnt-tracked), with safe fallback.
#if defined(__HIP_DEVICE_COMPILE__) && \
    __has_builtin(__builtin_amdgcn_global_load_async_to_lds_b128) && \
    __has_builtin(__builtin_amdgcn_s_wait_asynccnt)
#define USE_ASYNC_LDS 1
typedef __attribute__((address_space(1))) v4i glob_v4i;   // global int4 (b128 granule)
typedef __attribute__((address_space(3))) v4i lds_v4i;    // LDS int4 (b128 granule)
#else
#define USE_ASYNC_LDS 0
#endif

// ---------------- f32 -> bf16 convert ----------------
__global__ void cvt_f32_bf16(const float* __restrict__ s, bf16* __restrict__ d, int n) {
  int i = blockIdx.x * blockDim.x + threadIdx.x;
  int str = gridDim.x * blockDim.x;
  for (; i < n; i += str) d[i] = (bf16)s[i];
}

// ---------------- f32 (rows x cols) -> bf16 (rows x padcols), zero-padded ----------------
__global__ void cvt_f32_bf16_pad(const float* __restrict__ s, bf16* __restrict__ d,
                                 int rows, int cols, int padcols) {
  int i = blockIdx.x * blockDim.x + threadIdx.x;
  int str = gridDim.x * blockDim.x;
  int n = rows * padcols;
  for (; i < n; i += str) {
    int r = i / padcols, c = i - r * padcols;
    d[i] = (c < cols) ? (bf16)s[(size_t)r * cols + c] : (bf16)0.f;
  }
}

// ---------------- WMMA GEMM: C[M,N](f32) = A[M,K](bf16,rm) * B[K,N](bf16,rm) ----------------
// REQUIRES: M % 64 == 0, N % 128 == 0, K % 32 == 0 (guaranteed by padding).
// Block: 256 threads (8 waves, wave32). Tile 64(M) x 128(N), K-step 32, double-buffered LDS:
//   A staged row-major via async-to-LDS (ASYNCcnt); B staged TRANSPOSED (Bt[n][k]) via
//   early global b128 loads + deferred packed ds_store_b32. All fragments consumed as
//   contiguous ds_load_b128 pairs; all 5 fragments loaded before the 4-WMMA chain so a
//   single s_wait_dscnt covers them.
#define AS_LD 40    // 64 x 32 tile, row pitch 40 bf16 = 80B (16B-aligned rows)
#define BT_LD 40    // 128 x 32 transposed tile, row pitch 40 bf16 = 80B (16B-aligned rows)
__global__ __launch_bounds__(256)
void gemm_bf16f32(const bf16* __restrict__ A, const bf16* __restrict__ B,
                  float* __restrict__ C, int M, int N, int K)
{
  __shared__ __align__(16) bf16 As[2][64 * AS_LD];
  __shared__ __align__(16) bf16 Bt[2][128 * BT_LD];

  const int tid  = threadIdx.x;
  const int lane = tid & 31;
  const int wave = tid >> 5;
  const int wm   = wave & 3;   // wave row group: rows [16*wm, 16*wm+16)
  const int wn   = wave >> 2;  // wave col group: cols [64*wn, 64*wn+64)
  const int half = lane >> 4;
  const int l16  = lane & 15;
  const int m0   = blockIdx.y * 64;
  const int n0   = blockIdx.x * 128;

  v8f acc[4];
#pragma unroll
  for (int i = 0; i < 4; ++i)
#pragma unroll
    for (int j = 0; j < 8; ++j) acc[i][j] = 0.f;

  // Per-thread staging coordinates.
  const int arow = tid >> 2;             // A: 64 rows x 4 granules of 8 bf16
  const int acol = (tid & 3) << 3;
  const int bk2  = tid >> 4;             // B: k-pair index 0..15 (k = 2*bk2)
  const int bn8  = (tid & 15) << 3;      // B: n group base 0,8,...,120

  uint4 breg0, breg1;                    // in-flight B rows (k, k+1) x 8 n-values

  // ---- prologue: stage tile 0 into buffer 0 ----
  {
    const bf16* gpa = A + (size_t)(m0 + arow) * K + acol;
    bf16* lpa = &As[0][arow * AS_LD + acol];
#if USE_ASYNC_LDS
    __builtin_amdgcn_global_load_async_to_lds_b128(
        (glob_v4i*)const_cast<bf16*>(gpa), (lds_v4i*)lpa, 0, 0);
#else
    *(uint4*)lpa = *(const uint4*)gpa;
#endif
    breg0 = *(const uint4*)(B + (size_t)(2 * bk2) * N + n0 + bn8);
    breg1 = *(const uint4*)(B + (size_t)(2 * bk2 + 1) * N + n0 + bn8);
    const unsigned short* r0 = (const unsigned short*)&breg0;
    const unsigned short* r1 = (const unsigned short*)&breg1;
#pragma unroll
    for (int j = 0; j < 8; ++j) {
      unsigned int pk = (unsigned int)r0[j] | ((unsigned int)r1[j] << 16);
      *(unsigned int*)&Bt[0][(bn8 + j) * BT_LD + 2 * bk2] = pk;
    }
#if USE_ASYNC_LDS
    __builtin_amdgcn_s_wait_asynccnt(0);
#endif
    __syncthreads();
  }

  int cur = 0;
  for (int kt = 0; kt < K; kt += 32) {
    const int nxt = kt + 32;
    const bool has_next = nxt < K;

    // ---- issue next tile's loads early (overlap with WMMA compute) ----
    if (has_next) {
      const bf16* gpa = A + (size_t)(m0 + arow) * K + nxt + acol;
      bf16* lpa = &As[cur ^ 1][arow * AS_LD + acol];
#if USE_ASYNC_LDS
      __builtin_amdgcn_global_load_async_to_lds_b128(
          (glob_v4i*)const_cast<bf16*>(gpa), (lds_v4i*)lpa, 0, 0);
#else
      *(uint4*)lpa = *(const uint4*)gpa;
#endif
      breg0 = *(const uint4*)(B + (size_t)(nxt + 2 * bk2) * N + n0 + bn8);
      breg1 = *(const uint4*)(B + (size_t)(nxt + 2 * bk2 + 1) * N + n0 + bn8);
    }

    // ---- load ALL fragments first (10x ds_load_b128, one dscnt wait), then 4 WMMAs ----
    // A fragment: af[e] = K (kt + half*16 + e), contiguous in As row
    v16bf af;
    {
      const bf16* ap = &As[cur][(wm * 16 + l16) * AS_LD];
      ((uint4*)&af)[0] = *(const uint4*)(ap + half * 16);
      ((uint4*)&af)[1] = *(const uint4*)(ap + half * 16 + 8);
    }
    v16bf bfv[4];
#pragma unroll
    for (int nt = 0; nt < 4; ++nt) {
      // B fragment: bfv[nt][e] = K (kt + half*16 + e), contiguous in Bt row (n fixed)
      const bf16* bp = &Bt[cur][(wn * 64 + nt * 16 + l16) * BT_LD + half * 16];
      ((uint4*)&bfv[nt])[0] = ((const uint4*)bp)[0];
      ((uint4*)&bfv[nt])[1] = ((const uint4*)bp)[1];
    }
#pragma unroll
    for (int nt = 0; nt < 4; ++nt)
      acc[nt] = __builtin_amdgcn_wmma_f32_16x16x32_bf16(
          false, af, false, bfv[nt], (short)0, acc[nt], false, false);

    // ---- finish staging next tile: pack (k,k+1) pairs, transposed b32 stores ----
    if (has_next) {
      const unsigned short* r0 = (const unsigned short*)&breg0;
      const unsigned short* r1 = (const unsigned short*)&breg1;
#pragma unroll
      for (int j = 0; j < 8; ++j) {
        unsigned int pk = (unsigned int)r0[j] | ((unsigned int)r1[j] << 16);
        *(unsigned int*)&Bt[cur ^ 1][(bn8 + j) * BT_LD + 2 * bk2] = pk;
      }
#if USE_ASYNC_LDS
      __builtin_amdgcn_s_wait_asynccnt(0);
#endif
    }
    __syncthreads();
    cur ^= 1;
  }

  // Epilogue: C/D layout -> VGPR r holds M = r + half*8, N = l16. No guards (dims padded).
#pragma unroll
  for (int nt = 0; nt < 4; ++nt) {
    int gn = n0 + wn * 64 + nt * 16 + l16;
#pragma unroll
    for (int r = 0; r < 8; ++r) {
      int gm = m0 + wm * 16 + half * 8 + r;
      C[(size_t)gm * N + gn] = acc[nt][r];
    }
  }
}

// ---------------- RMS norm over a row slice, bf16 out ----------------
__global__ __launch_bounds__(256)
void rms_to_bf16(const float* __restrict__ in, int ld, int cols,
                 const float* __restrict__ w, bf16* __restrict__ out)
{
  __shared__ float red[256];
  const int r = blockIdx.x;
  const float* row = in + (size_t)r * ld;
  float s = 0.f;
  for (int c = threadIdx.x; c < cols; c += 256) { float v = row[c]; s += v * v; }
  red[threadIdx.x] = s;
  __syncthreads();
  for (int o = 128; o > 0; o >>= 1) {
    if (threadIdx.x < o) red[threadIdx.x] += red[threadIdx.x + o];
    __syncthreads();
  }
  const float scale = rsqrtf(red[0] / (float)cols + 1e-5f);
  for (int c = threadIdx.x; c < cols; c += 256)
    out[(size_t)r * cols + c] = (bf16)(row[c] * scale * w[c]);
}

// ---------------- elementwise log_sigmoid(x + bias) in place ----------------
__global__ void logsigmoid_bias(float* __restrict__ g, const float* __restrict__ b,
                                int n, int cols)
{
  int i = blockIdx.x * blockDim.x + threadIdx.x;
  int str = gridDim.x * blockDim.x;
  for (; i < n; i += str) {
    float x = g[i] + b[i % cols];
    g[i] = fminf(x, 0.f) - log1pf(__expf(-fabsf(x)));
  }
}

// ---------------- beta = sigmoid(x @ wb) : (T,H) ----------------
__global__ void beta_kernel(const float* __restrict__ x, const float* __restrict__ wb,
                            float* __restrict__ beta)
{
  int i = blockIdx.x * blockDim.x + threadIdx.x;
  if (i >= T_ * H_) return;
  int t = i / H_, h = i % H_;
  const float* xr = x + (size_t)t * DIM_;
  float s = 0.f;
  for (int k = 0; k < DIM_; ++k) s += xr[k] * wb[k * H_ + h];
  beta[i] = 1.f / (1.f + __expf(-s));
}

// ---------------- KDA sequential scan: one block per head, state in VGPRs ----------------
// 256 threads: thread (hf = tid>>7, col = tid&127) owns S[k, col] for k in [hf*96, hf*96+96).
__global__ __launch_bounds__(256)
void kda_scan(const float* __restrict__ qf,    // (T, H*DK)
              const float* __restrict__ kvb,   // (T, H*(DNOPE+DV)) = (T, 2048)
              const float* __restrict__ kva,   // (T, KVA_N); k_pe at col KVLR_ (512..575)
              const float* __restrict__ g,     // (T, H*DK)
              const float* __restrict__ beta,  // (T, H)
              bf16* __restrict__ ob)           // (T, H*DV) bf16
{
  __shared__ float sq[DK_], sk[DK_], se[DK_], sv[DV_];
  __shared__ float spart[2][DV_], sout[2][DV_], sred[64], sfac[2];
  const int h   = blockIdx.x;
  const int tid = threadIdx.x;
  const int col = tid & 127;
  const int hf  = tid >> 7;

  float S[96];
#pragma unroll
  for (int i = 0; i < 96; ++i) S[i] = 0.f;

  for (int t = 0; t < T_; ++t) {
    __syncthreads();  // protects sout/shared vecs from previous iteration
    if (tid < DK_) {
      sq[tid] = qf[(size_t)t * (H_ * DK_) + h * DK_ + tid];
      sk[tid] = (tid < DNOPE_) ? kvb[(size_t)t * 2048 + h * 256 + tid]
                               : kva[(size_t)t * KVA_N + KVLR_ + (tid - DNOPE_)];
      se[tid] = __expf(g[(size_t)t * (H_ * DK_) + h * DK_ + tid]);
    }
    if (tid < DV_) sv[tid] = kvb[(size_t)t * 2048 + h * 256 + DNOPE_ + tid];
    __syncthreads();

    // l2-norm factors for q and k (sum of squares over DK=192)
    if (tid < 32) {
      float aq = 0.f, ak = 0.f;
#pragma unroll
      for (int i = 0; i < 6; ++i) {
        float a = sq[tid * 6 + i], b = sk[tid * 6 + i];
        aq += a * a; ak += b * b;
      }
      sred[tid] = aq; sred[32 + tid] = ak;
    }
    __syncthreads();
    if (tid == 0) {
      float aq = 0.f, ak = 0.f;
      for (int i = 0; i < 32; ++i) { aq += sred[i]; ak += sred[32 + i]; }
      sfac[0] = rsqrtf(aq + 1e-6f) * 0.07216878364870323f;  // * DK^-0.5
      sfac[1] = rsqrtf(ak + 1e-6f);
    }
    __syncthreads();
    if (tid < DK_) { sq[tid] *= sfac[0]; sk[tid] *= sfac[1]; }
    __syncthreads();

    const float bt = beta[t * H_ + h];

    // pass 1: decay state, accumulate k^T S for this column
    float acc = 0.f;
#pragma unroll
    for (int i = 0; i < 96; ++i) {
      int k = hf * 96 + i;
      float s = S[i] * se[k];
      S[i] = s;
      acc += sk[k] * s;
    }
    spart[hf][col] = acc;
    __syncthreads();
    const float u = bt * (sv[col] - (spart[0][col] + spart[1][col]));

    // pass 2: rank-1 update, accumulate q^T S
    float acco = 0.f;
#pragma unroll
    for (int i = 0; i < 96; ++i) {
      int k = hf * 96 + i;
      float s = S[i] + sk[k] * u;
      S[i] = s;
      acco += sq[k] * s;
    }
    sout[hf][col] = acco;
    __syncthreads();
    if (hf == 0) {
      float o = sout[0][col] + sout[1][col];
      if (o != o) o = 0.f;
      else if (__builtin_isinf(o)) o = (o > 0.f) ? 1e4f : -1e4f;
      ob[(size_t)t * (H_ * DV_) + h * DV_ + col] = (bf16)o;
    }
  }
}

// ---------------- orchestration ----------------
extern "C" void kernel_launch(void* const* d_in, const int* in_sizes, int n_in,
                              void* d_out, int out_size, void* d_ws, size_t ws_size,
                              hipStream_t stream)
{
  (void)in_sizes; (void)n_in; (void)out_size; (void)ws_size;
  const float* x     = (const float*)d_in[0];
  // d_in[1]=cos, d_in[2]=sin unused by reference
  const float* wq_a  = (const float*)d_in[3];
  const float* qnw   = (const float*)d_in[4];
  const float* wq_b  = (const float*)d_in[5];
  const float* wkv_a = (const float*)d_in[6];
  const float* kvnw  = (const float*)d_in[7];
  const float* wkv_b = (const float*)d_in[8];
  const float* wg_w  = (const float*)d_in[9];
  const float* wg_b  = (const float*)d_in[10];
  const float* wb    = (const float*)d_in[11];
  const float* wo    = (const float*)d_in[12];
  float* out = (float*)d_out;

  char* p = (char*)d_ws;
  auto alloc = [&](size_t bytes) -> char* {
    char* r = p;
    p += (bytes + 255) & ~(size_t)255;
    return r;
  };
  bf16*  xb     = (bf16*) alloc((size_t)DIM_ * T_ * 2);
  bf16*  wqa_b  = (bf16*) alloc((size_t)DIM_ * QLR_ * 2);
  bf16*  wqb_b  = (bf16*) alloc((size_t)QLR_ * (H_ * DK_) * 2);
  bf16*  wkva_b = (bf16*) alloc((size_t)DIM_ * KVA_N * 2);        // zero-padded to 640 cols
  bf16*  wkvb_b = (bf16*) alloc((size_t)KVLR_ * (H_ * 256) * 2);
  bf16*  wgw_b  = (bf16*) alloc((size_t)DIM_ * (H_ * DK_) * 2);
  bf16*  wo_b   = (bf16*) alloc((size_t)(H_ * DV_) * DIM_ * 2);
  float* qlat_f = (float*)alloc((size_t)T_ * QLR_ * 4);
  bf16*  qlat_b = (bf16*) alloc((size_t)T_ * QLR_ * 2);
  float* qfull  = (float*)alloc((size_t)T_ * (H_ * DK_) * 4);
  float* kva_f  = (float*)alloc((size_t)T_ * KVA_N * 4);
  bf16*  kvc_b  = (bf16*) alloc((size_t)T_ * KVLR_ * 2);
  float* kv_f   = (float*)alloc((size_t)T_ * (H_ * 256) * 4);
  float* g_f    = (float*)alloc((size_t)T_ * (H_ * DK_) * 4);
  float* beta_f = (float*)alloc((size_t)T_ * H_ * 4);
  bf16*  ob_b   = (bf16*) alloc((size_t)T_ * (H_ * DV_) * 2);

  const int CB = 256;
  // bf16 conversions of activations + weights (wkv_a zero-padded 576->640 cols)
  cvt_f32_bf16<<<512, CB, 0, stream>>>(x,     xb,     DIM_ * T_);
  cvt_f32_bf16<<<512, CB, 0, stream>>>(wq_a,  wqa_b,  DIM_ * QLR_);
  cvt_f32_bf16<<<512, CB, 0, stream>>>(wq_b,  wqb_b,  QLR_ * H_ * DK_);
  cvt_f32_bf16_pad<<<512, CB, 0, stream>>>(wkv_a, wkva_b, DIM_, KVLR_ + DROPE_, KVA_N);
  cvt_f32_bf16<<<512, CB, 0, stream>>>(wkv_b, wkvb_b, KVLR_ * H_ * 256);
  cvt_f32_bf16<<<512, CB, 0, stream>>>(wg_w,  wgw_b,  DIM_ * H_ * DK_);
  cvt_f32_bf16<<<512, CB, 0, stream>>>(wo,    wo_b,   H_ * DV_ * DIM_);

  // q_lat = x @ wq_a ; rms ; q = q_lat @ wq_b
  gemm_bf16f32<<<dim3(QLR_ / 128, T_ / 64), 256, 0, stream>>>(xb, wqa_b, qlat_f, T_, QLR_, DIM_);
  rms_to_bf16<<<T_, 256, 0, stream>>>(qlat_f, QLR_, QLR_, qnw, qlat_b);
  gemm_bf16f32<<<dim3((H_ * DK_) / 128, T_ / 64), 256, 0, stream>>>(qlat_b, wqb_b, qfull, T_, H_ * DK_, QLR_);

  // kv_all = x @ wkv_a(padded) ; rms first 512 ; kv = kv_c @ wkv_b
  gemm_bf16f32<<<dim3(KVA_N / 128, T_ / 64), 256, 0, stream>>>(xb, wkva_b, kva_f, T_, KVA_N, DIM_);
  rms_to_bf16<<<T_, 256, 0, stream>>>(kva_f, KVA_N, KVLR_, kvnw, kvc_b);
  gemm_bf16f32<<<dim3((H_ * 256) / 128, T_ / 64), 256, 0, stream>>>(kvc_b, wkvb_b, kv_f, T_, H_ * 256, KVLR_);

  // g = log_sigmoid(x @ wg_w + wg_b)
  gemm_bf16f32<<<dim3((H_ * DK_) / 128, T_ / 64), 256, 0, stream>>>(xb, wgw_b, g_f, T_, H_ * DK_, DIM_);
  logsigmoid_bias<<<1024, CB, 0, stream>>>(g_f, wg_b, T_ * H_ * DK_, H_ * DK_);

  // beta = sigmoid(x @ wb)
  beta_kernel<<<(T_ * H_ + CB - 1) / CB, CB, 0, stream>>>(x, wb, beta_f);

  // sequential delta-rule scan, one block per head
  kda_scan<<<H_, 256, 0, stream>>>(qfull, kv_f, kva_f, g_f, beta_f, ob_b);

  // out = o @ wo (f32 result directly into d_out)
  gemm_bf16f32<<<dim3(DIM_ / 128, T_ / 64), 256, 0, stream>>>(ob_b, wo_b, out, T_, DIM_, H_ * DV_);
}
